// MultiHead2DAttention_v2_26766236189174
// MI455X (gfx1250) — matrix operand order repaired and checked by reading, not compiled, not run on hardware
//
#include <hip/hip_runtime.h>
#include <stdint.h>

// ---------------------------------------------------------------------------
// Relative-position multi-head attention for MI455X (gfx1250), bf16 WMMA.
//   B=4, L=1024, D=1024, H=16, DH=64, MAXREL=64  (J = 2*64+1 = 129 buckets)
// Pipeline:
//   1) convert x -> bf16, W* -> bf16 transposed, rel tables -> padded bf16
//   2) Q/K/V projection GEMMs (WMMA bf16, fp32 acc), Q pre-scaled by DH^-1/2,
//      V stored transposed per head [b,h,d,k]
//   3) Rk[b,h,q,j] = Q . rel_k^T   (WMMA GEMM, j padded 129->144)
//   4) fused attention: two-pass softmax, logits = QK^T (WMMA) + Rk gather,
//      O = P@V (WMMA) + S@rel_v (WMMA), S built with LDS ds_add_f32 atomics.
//      K/V tiles staged in LDS with GLOBAL_LOAD_ASYNC_TO_LDS_B128 (ASYNCcnt),
//      double-buffered, shared by all 4 waves.
//   5) out = O @ Wo (WMMA, fp32 store)
// ---------------------------------------------------------------------------

typedef __bf16 bf16;
typedef __attribute__((ext_vector_type(16))) __bf16 v16bf;
typedef __attribute__((ext_vector_type(8)))  __bf16 v8bf;
typedef __attribute__((ext_vector_type(8)))  float  v8f;

#define DEV __device__ __forceinline__

DEV int lane_id() { return (int)(threadIdx.x & 31); }

DEV v8f wmma_bf16(v16bf a, v16bf b, v8f c) {
  return __builtin_amdgcn_wmma_f32_16x16x32_bf16(false, a, false, b, (short)0,
                                                 c, false, false);
}

// A fragment (16x32, bf16) from row-major [.., ld] at (m0, k0).
// gfx1250 layout: lane l -> M = l&15 ; lanes<16: K in {k0..k0+7, k0+16..k0+23}
//                                      lanes>=16: +8.  Two 16B contiguous loads.
DEV v16bf load_a_g(const bf16* __restrict__ A, int ld, int m0, int k0) {
  int l = lane_id();
  const bf16* p = A + (size_t)(m0 + (l & 15)) * ld + (k0 + ((l >> 4) << 3));
  v16bf a;
  ((v8bf*)&a)[0] = *(const v8bf*)p;        // K = base .. base+7
  ((v8bf*)&a)[1] = *(const v8bf*)(p + 16); // K = base+16 .. base+23
  return a;
}

// B fragment (32x16 KxN, bf16) from "Bt" storage Bt[n][k] row-major (ld over k).
// gfx1250 layout: lane l -> N = n0 + (l&15); K in {k0..k0+7, k0+16..k0+23} (+8 for l>=16). Two 16B loads.
// Works for both global and LDS (generic pointer -> ds_load for __shared__).
DEV v16bf load_b_any(const bf16* Bt, int ld, int n0, int k0) {
  int l = lane_id();
  const bf16* p = Bt + (size_t)(n0 + (l & 15)) * ld + (k0 + ((l >> 4) << 3));
  v16bf b;
  ((v8bf*)&b)[0] = *(const v8bf*)p;
  ((v8bf*)&b)[1] = *(const v8bf*)(p + 16);
  return b;
}

// A fragment built from an LDS f32 tile base[16][ld] (converted to bf16).
DEV v16bf lds_a_frag_f32(const float* base, int ld) {
  int l = lane_id();
  int m = l & 15;
  int kb = (l >> 4) << 3;
  v16bf a;
#pragma unroll
  for (int e = 0; e < 16; ++e) {
    int k = (e & 7) + ((e & 8) << 1) + kb; // {0..7,16..23} (+8 for hi lanes)
    a[e] = (bf16)base[m * ld + k];
  }
  return a;
}

// Async copy: 16 bytes per lane, global -> LDS (one wave-level instruction,
// tracked by ASYNCcnt).  lds_byte = byte offset within the block's LDS alloc
// (low 32 bits of a generic pointer to __shared__ per the LDS aperture rule).
DEV void async_ld_b128(uint32_t lds_byte, const void* gaddr) {
  asm volatile("global_load_async_to_lds_b128 %0, %1, off"
               :
               : "v"(lds_byte), "v"(gaddr)
               : "memory");
}

DEV void wait_asynccnt0() {
  asm volatile("s_wait_asynccnt 0" ::: "memory");
}

// ---------------------------------------------------------------------------
// Conversion / packing kernels
// ---------------------------------------------------------------------------

__global__ void cvt_bf16_kernel(const float* __restrict__ in,
                                bf16* __restrict__ out, int n) {
  int i = blockIdx.x * blockDim.x + threadIdx.x;
  if (i < n) out[i] = (bf16)in[i];
}

// Wt[n*1024 + k] = W[k*1024 + n]   (1024x1024)
__global__ void transpose_w_kernel(const float* __restrict__ W,
                                   bf16* __restrict__ Wt) {
  int i = blockIdx.x * blockDim.x + threadIdx.x;
  int n = i >> 10, k = i & 1023;
  Wt[i] = (bf16)W[k * 1024 + n];
}

// relk_bf [144][64] (j-major, zero padded j>=129)
// relv_t  [64][160] (d-major, zero padded j>=129)
__global__ void pack_rel_kernel(const float* __restrict__ rk,
                                const float* __restrict__ rv,
                                bf16* __restrict__ relk_bf,
                                bf16* __restrict__ relv_t) {
  int i = blockIdx.x * blockDim.x + threadIdx.x;
  if (i < 144 * 64) {
    int j = i >> 6, d = i & 63;
    relk_bf[i] = (bf16)((j < 129) ? rk[j * 64 + d] : 0.0f);
  } else if (i < 144 * 64 + 64 * 160) {
    int i2 = i - 144 * 64;
    int d = i2 / 160, j = i2 % 160;
    relv_t[i2] = (bf16)((j < 129) ? rv[j * 64 + d] : 0.0f);
  }
}

// ---------------------------------------------------------------------------
// Generic bf16 WMMA GEMM:  C[M x 1024] = A[M x 1024] @ Bt^T  (Bt is [n][k])
// mode 0: Q  -> bf16 [b,h,seq,dh], scaled
// mode 1: K  -> bf16 [b,h,seq,dh]
// mode 2: V  -> bf16 [b,h,dh,seq]  (transposed per head)
// mode 3: f32 row-major [row][col] (final output)
// Block: 256 thr = 8 waves; wave tile 32x32 (2x2 WMMA); block tile 128x64.
// ---------------------------------------------------------------------------

DEV void store_tile(v8f c, int m0, int n0, int mode, bf16* __restrict__ outb,
                    float* __restrict__ outf, float scale) {
  int l = lane_id();
  int g = l >> 4;
  int nc = l & 15;
#pragma unroll
  for (int r = 0; r < 8; ++r) {
    int row = m0 + r + 8 * g;  // 0..4095  = b*1024 + seq
    int col = n0 + nc;         // 0..1023  = h*64 + dh
    float v = c[r] * scale;
    int b = row >> 10, seq = row & 1023;
    int h = col >> 6, dh = col & 63;
    if (mode <= 1) {
      outb[(((size_t)(b * 16 + h)) * 1024 + seq) * 64 + dh] = (bf16)v;
    } else if (mode == 2) {
      outb[(((size_t)(b * 16 + h)) * 64 + dh) * 1024 + seq] = (bf16)v;
    } else {
      outf[(size_t)row * 1024 + col] = v;
    }
  }
}

__global__ __launch_bounds__(256) void gemm_bf16_wmma(
    const bf16* __restrict__ A, const bf16* __restrict__ Bt,
    bf16* __restrict__ outb, float* __restrict__ outf, int mode, float scale) {
  int w = threadIdx.x >> 5;
  int m0 = blockIdx.x * 128 + (w >> 1) * 32;
  int n0 = blockIdx.y * 64 + (w & 1) * 32;
  v8f c00 = {}, c01 = {}, c10 = {}, c11 = {};
#pragma unroll 4
  for (int k0 = 0; k0 < 1024; k0 += 32) {
    __builtin_prefetch(A + (size_t)m0 * 1024 + k0 + 256, 0, 1);
    v16bf a0 = load_a_g(A, 1024, m0, k0);
    v16bf a1 = load_a_g(A, 1024, m0 + 16, k0);
    v16bf b0 = load_b_any(Bt, 1024, n0, k0);
    v16bf b1 = load_b_any(Bt, 1024, n0 + 16, k0);
    c00 = wmma_bf16(a0, b0, c00);
    c01 = wmma_bf16(a0, b1, c01);
    c10 = wmma_bf16(a1, b0, c10);
    c11 = wmma_bf16(a1, b1, c11);
  }
  store_tile(c00, m0, n0, mode, outb, outf, scale);
  store_tile(c01, m0, n0 + 16, mode, outb, outf, scale);
  store_tile(c10, m0 + 16, n0, mode, outb, outf, scale);
  store_tile(c11, m0 + 16, n0 + 16, mode, outb, outf, scale);
}

// ---------------------------------------------------------------------------
// Rk[b,h,q,j] = Q[b,h,q,:] . rel_k[j,:]   j in [0,144), f32
// Block: 128 thr = 4 waves; wave = one 16-row q tile; 9 j-tiles x 2 k-steps.
// ---------------------------------------------------------------------------

__global__ __launch_bounds__(128) void rk_gemm_kernel(
    const bf16* __restrict__ Q, const bf16* __restrict__ relk_bf,
    float* __restrict__ Rk) {
  int bh = blockIdx.x;                     // 0..63
  int w = threadIdx.x >> 5;
  int l = lane_id();
  int q0 = blockIdx.y * 64 + w * 16;       // 0..1008
  const bf16* Qh = Q + (size_t)bh * 1024 * 64;
  v16bf a0 = load_a_g(Qh, 64, q0, 0);
  v16bf a1 = load_a_g(Qh, 64, q0, 32);
#pragma unroll
  for (int nt = 0; nt < 9; ++nt) {
    v16bf b0 = load_b_any(relk_bf, 64, nt * 16, 0);
    v16bf b1 = load_b_any(relk_bf, 64, nt * 16, 32);
    v8f c = {};
    c = wmma_bf16(a0, b0, c);
    c = wmma_bf16(a1, b1, c);
    int g = l >> 4, nc = l & 15;
#pragma unroll
    for (int r = 0; r < 8; ++r) {
      int qg = q0 + r + 8 * g;
      Rk[((size_t)bh * 1024 + qg) * 144 + nt * 16 + nc] = c[r];
    }
  }
}

// ---------------------------------------------------------------------------
// Fused attention. Grid (bh=64, qb=16). Block 128 thr = 4 waves.
// Wave w owns q rows [qb*64 + w*16, +16).  Two-pass softmax over 1024 keys.
// K tiles (16x64 bf16 = 2KB) and V pair-tiles (64x32 bf16 = 4KB) are staged
// into LDS with async global->LDS copies, double-buffered, shared by 4 waves.
// ---------------------------------------------------------------------------

__global__ __launch_bounds__(128) void attn_kernel(
    const bf16* __restrict__ Q, const bf16* __restrict__ K,
    const bf16* __restrict__ Vt, const float* __restrict__ Rk,
    const bf16* __restrict__ relv_t, bf16* __restrict__ O) {
  int bh = blockIdx.x;
  int qb = blockIdx.y;
  int w = threadIdx.x >> 5;
  int t = threadIdx.x;
  int l = lane_id();
  int g = l >> 4;
  int ncol = l & 15;
  int q0 = qb * 64 + w * 16;

  const bf16* Qh = Q + (size_t)bh * 1024 * 64;
  const bf16* Kh = K + (size_t)bh * 1024 * 64;
  const bf16* Vh = Vt + (size_t)bh * 64 * 1024;
  const float* Rkh = Rk + (size_t)bh * 1024 * 144;

  __shared__ __align__(32) float Sld[4][16][160];  // rel-bucket accumulators
  __shared__ __align__(32) float Pst[4][16][32];   // P staging C-frag->A-frag
  __shared__ __align__(32) bf16 Ktile[2][16][64];  // K double buffer (2KB ea)
  __shared__ __align__(32) bf16 Vtile[2][64][32];  // V double buffer (4KB ea)

  for (int i = l; i < 16 * 160; i += 32) ((float*)Sld[w])[i] = 0.0f;

  // per-thread async copy slots (block-cooperative; 1 K chunk, 2 V chunks)
  uint32_t ktile_lds0 = (uint32_t)(uintptr_t)&Ktile[0][0][0];
  uint32_t vtile_lds0 = (uint32_t)(uintptr_t)&Vtile[0][0][0];
  auto issueK = [&](int kt, int buf) {
    // 16 rows x 64 bf16 contiguous = 2048B; chunk t = 16B
    async_ld_b128(ktile_lds0 + (uint32_t)buf * 2048u + (uint32_t)t * 16u,
                  Kh + (size_t)kt * 1024 + t * 8);
  };
  auto issueV = [&](int p, int buf) {
    // rows d=0..63 of Vt, cols [p*32, p*32+32): 256 chunks of 16B, 2/thread
#pragma unroll
    for (int i = 0; i < 2; ++i) {
      int c = t + i * 128;
      int row = c >> 2, cc = c & 3;
      async_ld_b128(vtile_lds0 + (uint32_t)buf * 4096u +
                        (uint32_t)(row * 64 + cc * 16),
                    Vh + (size_t)row * 1024 + p * 32 + cc * 8);
    }
  };

  v16bf aq0 = load_a_g(Qh, 64, q0, 0);
  v16bf aq1 = load_a_g(Qh, 64, q0, 32);

  float mrow[8];
#pragma unroll
  for (int r = 0; r < 8; ++r) mrow[r] = -1e30f;

  // ---- pass 1: row maxima of (QK^T + Rk-gather) ----
  issueK(0, 0);
  wait_asynccnt0();
  __syncthreads();
  for (int kt = 0; kt < 64; ++kt) {
    if (kt < 63) issueK(kt + 1, (kt + 1) & 1);
    v16bf b0 = load_b_any(&Ktile[kt & 1][0][0], 64, 0, 0);
    v16bf b1 = load_b_any(&Ktile[kt & 1][0][0], 64, 0, 32);
    v8f c = {};
    c = wmma_bf16(aq0, b0, c);
    c = wmma_bf16(aq1, b1, c);
    int kg = kt * 16 + ncol;
#pragma unroll
    for (int r = 0; r < 8; ++r) {
      int qg = q0 + r + 8 * g;
      int dd = kg - qg;
      dd = dd < -64 ? -64 : (dd > 64 ? 64 : dd);
      float v = c[r] + Rkh[(size_t)qg * 144 + dd + 64];
      v = fmaxf(v, __shfl_xor(v, 1, 32));
      v = fmaxf(v, __shfl_xor(v, 2, 32));
      v = fmaxf(v, __shfl_xor(v, 4, 32));
      v = fmaxf(v, __shfl_xor(v, 8, 32));
      mrow[r] = fmaxf(mrow[r], v);
    }
    if (kt < 63) {
      wait_asynccnt0();  // next K tile landed (overlapped with compute above)
      __syncthreads();
    }
  }
  __syncthreads();

  // ---- pass 2: unnormalized exp, O += P@V (WMMA), S bucket scatter ----
  v8f oacc[4];
#pragma unroll
  for (int nt = 0; nt < 4; ++nt) oacc[nt] = (v8f){};
  float ssum[8];
#pragma unroll
  for (int r = 0; r < 8; ++r) ssum[r] = 0.0f;

  issueK(0, 0);
  wait_asynccnt0();
  __syncthreads();
#pragma unroll 2
  for (int kt = 0; kt < 64; ++kt) {
    if (kt < 63) issueK(kt + 1, (kt + 1) & 1);
    if ((kt & 1) == 0) issueV(kt >> 1, (kt >> 1) & 1);  // V pair for kt, kt+1
    v16bf b0 = load_b_any(&Ktile[kt & 1][0][0], 64, 0, 0);
    v16bf b1 = load_b_any(&Ktile[kt & 1][0][0], 64, 0, 32);
    v8f c = {};
    c = wmma_bf16(aq0, b0, c);
    c = wmma_bf16(aq1, b1, c);
    int kg = kt * 16 + ncol;
#pragma unroll
    for (int r = 0; r < 8; ++r) {
      int qg = q0 + r + 8 * g;
      int dd = kg - qg;
      dd = dd < -64 ? -64 : (dd > 64 ? 64 : dd);
      float lg = c[r] + Rkh[(size_t)qg * 144 + dd + 64];
      float pr = __expf(lg - mrow[r]);
      ssum[r] += pr;
      atomicAdd(&Sld[w][r + 8 * g][dd + 64], pr);      // ds_add_f32
      Pst[w][r + 8 * g][((kt & 1) << 4) + ncol] = pr;  // stage for A-frag
    }
    if (kt & 1) {
      // wave-private LDS RAW: DS ops are in-order per wave, just drain DScnt
      asm volatile("s_wait_dscnt 0" ::: "memory");
      v16bf ap = lds_a_frag_f32(&Pst[w][0][0], 32);
      int vb = (kt >> 1) & 1;
#pragma unroll
      for (int nt = 0; nt < 4; ++nt) {
        v16bf bv = load_b_any(&Vtile[vb][0][0], 32, nt * 16, 0);
        oacc[nt] = wmma_bf16(ap, bv, oacc[nt]);
      }
    }
    wait_asynccnt0();  // staged-ahead K/V landed (overlapped with compute)
    __syncthreads();
  }

  // ---- O += S @ rel_v  (K dim 160, zero-padded) ----
  asm volatile("s_wait_dscnt 0" ::: "memory");
#pragma unroll
  for (int ks = 0; ks < 5; ++ks) {
    v16bf as = lds_a_frag_f32(&Sld[w][0][ks * 32], 160);
#pragma unroll
    for (int nt = 0; nt < 4; ++nt) {
      v16bf bv = load_b_any(relv_t, 160, nt * 16, ks * 32);
      oacc[nt] = wmma_bf16(as, bv, oacc[nt]);
    }
  }

  // ---- normalize and store O as bf16 [b, seq, h, dh] ----
#pragma unroll
  for (int r = 0; r < 8; ++r) {
    ssum[r] += __shfl_xor(ssum[r], 1, 32);
    ssum[r] += __shfl_xor(ssum[r], 2, 32);
    ssum[r] += __shfl_xor(ssum[r], 4, 32);
    ssum[r] += __shfl_xor(ssum[r], 8, 32);
  }
  int bcol = (bh & 15) * 64;
  int brow0 = (bh >> 4) * 1024;
#pragma unroll
  for (int nt = 0; nt < 4; ++nt) {
#pragma unroll
    for (int r = 0; r < 8; ++r) {
      int qg = q0 + r + 8 * g;
      float inv = 1.0f / ssum[r];
      O[(size_t)(brow0 + qg) * 1024 + bcol + nt * 16 + ncol] =
          (bf16)(oacc[nt][r] * inv);
    }
  }
}

// ---------------------------------------------------------------------------
// Host launcher
// ---------------------------------------------------------------------------

extern "C" void kernel_launch(void* const* d_in, const int* in_sizes, int n_in,
                              void* d_out, int out_size, void* d_ws,
                              size_t ws_size, hipStream_t stream) {
  (void)in_sizes; (void)n_in; (void)out_size; (void)ws_size;
  const float* x    = (const float*)d_in[0];
  const float* Wq   = (const float*)d_in[1];
  const float* Wk   = (const float*)d_in[2];
  const float* Wv   = (const float*)d_in[3];
  const float* Wo   = (const float*)d_in[4];
  const float* relk = (const float*)d_in[5];
  const float* relv = (const float*)d_in[6];
  float* out = (float*)d_out;

  char* base = (char*)d_ws;
  size_t off = 0;
  auto take = [&](size_t bytes) -> char* {
    char* r = base + off;
    off += (bytes + 255) & ~(size_t)255;
    return r;
  };
  bf16* Xbf = (bf16*)take(4096ull * 1024 * 2);
  bf16* Wqt = (bf16*)take(1024ull * 1024 * 2);
  bf16* Wkt = (bf16*)take(1024ull * 1024 * 2);
  bf16* Wvt = (bf16*)take(1024ull * 1024 * 2);
  bf16* Wot = (bf16*)take(1024ull * 1024 * 2);
  bf16* Qbf = (bf16*)take(64ull * 1024 * 64 * 2);
  bf16* Kbf = (bf16*)take(64ull * 1024 * 64 * 2);
  bf16* Vtb = (bf16*)take(64ull * 64 * 1024 * 2);
  bf16* RKb = (bf16*)take(144ull * 64 * 2);
  bf16* RVt = (bf16*)take(64ull * 160 * 2);
  float* Rk = (float*)take(64ull * 1024 * 144 * 4);
  bf16* Obf = (bf16*)take(4096ull * 1024 * 2);

  cvt_bf16_kernel<<<dim3(4096 * 1024 / 256), 256, 0, stream>>>(x, Xbf,
                                                               4096 * 1024);
  transpose_w_kernel<<<dim3(1024 * 1024 / 256), 256, 0, stream>>>(Wq, Wqt);
  transpose_w_kernel<<<dim3(1024 * 1024 / 256), 256, 0, stream>>>(Wk, Wkt);
  transpose_w_kernel<<<dim3(1024 * 1024 / 256), 256, 0, stream>>>(Wv, Wvt);
  transpose_w_kernel<<<dim3(1024 * 1024 / 256), 256, 0, stream>>>(Wo, Wot);
  pack_rel_kernel<<<dim3((144 * 64 + 64 * 160 + 255) / 256), 256, 0, stream>>>(
      relk, relv, RKb, RVt);

  // Q (scaled by DH^-0.5 = 0.125), K, V(transposed) projections
  gemm_bf16_wmma<<<dim3(32, 16), 256, 0, stream>>>(Xbf, Wqt, Qbf, nullptr, 0,
                                                   0.125f);
  gemm_bf16_wmma<<<dim3(32, 16), 256, 0, stream>>>(Xbf, Wkt, Kbf, nullptr, 1,
                                                   1.0f);
  gemm_bf16_wmma<<<dim3(32, 16), 256, 0, stream>>>(Xbf, Wvt, Vtb, nullptr, 2,
                                                   1.0f);

  rk_gemm_kernel<<<dim3(64, 16), 128, 0, stream>>>(Qbf, RKb, Rk);
  attn_kernel<<<dim3(64, 16), 128, 0, stream>>>(Qbf, Kbf, Vtb, Rk, RVt, Obf);

  // final projection, f32 output
  gemm_bf16_wmma<<<dim3(32, 16), 256, 0, stream>>>(Obf, Wot, nullptr, out, 3,
                                                   1.0f);
}
